// SALTEdoraLinear_38568806318480
// MI455X (gfx1250) — matrix operand order repaired and checked by reading, not compiled, not run on hardware
//
#include <hip/hip_runtime.h>

typedef __attribute__((ext_vector_type(16))) __bf16 v16bf;
typedef __attribute__((ext_vector_type(8)))  __bf16 v8bf;
typedef __attribute__((ext_vector_type(8)))  float  v8f;

#define OUT_N 4096
#define IN_K  4096
#define RNK   64
#define MROWS 16384   // B*S = 4*4096

__device__ __forceinline__ unsigned short f32_to_bf16(float f) {
    union { float f; unsigned int u; } c; c.f = f;
    unsigned int lsb = (c.u >> 16) & 1u;
    c.u += 0x7fffu + lsb;                 // round-to-nearest-even
    return (unsigned short)(c.u >> 16);
}

// CDNA5 async global->LDS copy (per-lane 16B), tracked by ASYNCcnt.
__device__ __forceinline__ void async_copy_b128(unsigned lds_byte_off, const void* gaddr) {
    asm volatile("global_load_async_to_lds_b128 %0, %1, off"
                 :
                 : "v"(lds_byte_off), "v"(gaddr)
                 : "memory");
}
__device__ __forceinline__ void wait_async0() {
    asm volatile("s_wait_asynccnt 0x0" ::: "memory");
}
// Low 32 bits of a generic pointer to LDS == LDS byte address (aperture rule).
__device__ __forceinline__ unsigned lds_addr(const void* p) {
    return (unsigned)(unsigned long long)p;
}

// ---------------------------------------------------------------- sigma[r]
__global__ void k_sigma(const float* __restrict__ S_top, const float* __restrict__ alpha,
                        const float* __restrict__ beta, float* __restrict__ sigma) {
    int t = threadIdx.x;
    if (t < RNK) {
        float v = S_top[t] * alpha[t] + beta[t];
        sigma[t] = v > 0.f ? v : 0.f;
    }
}

// ---------------------------------------------------------------- RV = R @ Vh_tail   [64, 4096]
__global__ __launch_bounds__(256) void k_rv(const float* __restrict__ Rm,
                                            const float* __restrict__ Vh_tail,
                                            float* __restrict__ RV) {
    int r = blockIdx.y;
    int i = blockIdx.x * 256 + threadIdx.x;
    float acc = 0.f;
#pragma unroll 8
    for (int q = 0; q < RNK; ++q)
        acc = fmaf(Rm[r * RNK + q], Vh_tail[(size_t)q * IN_K + i], acc);
    RV[(size_t)r * IN_K + i] = acc;
}

// ---------------------------------------------------------------- W~ (bf16) = W + (U_top*sigma)@Vh_top + (U_tail*S_tail)@RV
__global__ __launch_bounds__(256) void k_wtilde(const float* __restrict__ W,
                                                const float* __restrict__ U_top,
                                                const float* __restrict__ U_tail,
                                                const float* __restrict__ S_tail,
                                                const float* __restrict__ sigma,
                                                const float* __restrict__ Vh_top,
                                                const float* __restrict__ RV,
                                                unsigned short* __restrict__ Wtb) {
    __shared__ float aS[16 * RNK];
    __shared__ float cS[16 * RNK];
    const int tid   = threadIdx.x;
    const int oBase = blockIdx.y * 16;
    const int i     = blockIdx.x * 256 + tid;
#pragma unroll
    for (int j = 0; j < 4; ++j) {
        int idx = tid + j * 256;
        int oo = idx >> 6, rr = idx & 63;
        aS[idx] = U_top[(size_t)(oBase + oo) * RNK + rr] * sigma[rr];
        cS[idx] = U_tail[(size_t)(oBase + oo) * RNK + rr] * S_tail[rr];
    }
    __syncthreads();
    float acc[16];
#pragma unroll
    for (int oo = 0; oo < 16; ++oo) acc[oo] = 0.f;
    for (int r = 0; r < RNK; ++r) {
        float v = Vh_top[(size_t)r * IN_K + i];
        float u = RV[(size_t)r * IN_K + i];
#pragma unroll
        for (int oo = 0; oo < 16; ++oo)
            acc[oo] = fmaf(aS[oo * RNK + r], v, fmaf(cS[oo * RNK + r], u, acc[oo]));
    }
#pragma unroll
    for (int oo = 0; oo < 16; ++oo) {
        size_t off = (size_t)(oBase + oo) * IN_K + i;
        Wtb[off] = f32_to_bf16(W[off] + acc[oo]);
    }
}

// ---------------------------------------------------------------- x (f32) -> xb (bf16), float4-vectorized
__global__ __launch_bounds__(256) void k_xconv(const float* __restrict__ x,
                                               unsigned short* __restrict__ xb) {
    int idx = blockIdx.x * 256 + threadIdx.x;   // one float4 per thread
    const float4 f = ((const float4*)x)[idx];
    ushort4 o;
    o.x = f32_to_bf16(f.x); o.y = f32_to_bf16(f.y);
    o.z = f32_to_bf16(f.z); o.w = f32_to_bf16(f.w);
    ((ushort4*)xb)[idx] = o;
}

// ---------------------------------------------------------------- main GEMM: out = xb @ Wtb^T + bias
// Block: 256 thr (8 waves), macro tile 128M x 128N, K-step 32.
// Double-buffered LDS filled by global_load_async_to_lds_b128 (ASYNCcnt).
// Wave grid 2(M) x 4(N); each wave: 4x2 tiles of 16x16 via v_wmma_f32_16x16x32_bf16.
#define LDS_STRIDE 40   // 32 halves + 8 pad (80B rows, 16B aligned)

union FragU { v16bf v; v8bf h[2]; };

__global__ __launch_bounds__(256) void k_gemm(const unsigned short* __restrict__ xb,
                                              const unsigned short* __restrict__ Wtb,
                                              const float* __restrict__ bias,
                                              float* __restrict__ out) {
    __shared__ unsigned short ldsA[2][128 * LDS_STRIDE];
    __shared__ unsigned short ldsB[2][128 * LDS_STRIDE];

    const int tid  = threadIdx.x;
    const int lane = tid & 31;
    const int wave = tid >> 5;
    const int wm   = wave >> 2;        // 0..1
    const int wn   = wave & 3;         // 0..3
    const int r    = lane & 15;
    const int h    = lane >> 4;

    const int mBase = blockIdx.y * 128;
    const int nBase = blockIdx.x * 128;

    v8f acc[4][2];
#pragma unroll
    for (int im = 0; im < 4; ++im)
#pragma unroll
        for (int in_ = 0; in_ < 2; ++in_)
#pragma unroll
            for (int e = 0; e < 8; ++e) acc[im][in_][e] = 0.f;

    // Per-thread copy slot: 512 16B-chunks per 128x32 tile; thread does rows {row, row+64}.
    const int row = tid >> 2;
    const int ko  = (tid & 3) * 8;
    const unsigned short* gA = xb  + (size_t)(mBase + row) * IN_K + ko;
    const unsigned short* gB = Wtb + (size_t)(nBase + row) * IN_K + ko;
    const unsigned aOff0 = lds_addr(&ldsA[0][row * LDS_STRIDE + ko]);
    const unsigned bOff0 = lds_addr(&ldsB[0][row * LDS_STRIDE + ko]);
    const unsigned bufB  = 128 * LDS_STRIDE * 2;   // bytes per buffer
    const unsigned rowJ  = 64 * LDS_STRIDE * 2;    // bytes for +64 rows

    auto issue = [&](unsigned bufOff, int kt) {
        const unsigned short* a = gA + kt * 32;
        const unsigned short* b = gB + kt * 32;
        const unsigned ao = aOff0 + bufOff;
        const unsigned bo = bOff0 + bufOff;
        async_copy_b128(ao,        a);
        async_copy_b128(ao + rowJ, a + (size_t)64 * IN_K);
        async_copy_b128(bo,        b);
        async_copy_b128(bo + rowJ, b + (size_t)64 * IN_K);
    };

    auto compute = [&](unsigned bufOff) {
        const unsigned short* lA = (const unsigned short*)((const char*)&ldsA[0][0] + bufOff);
        const unsigned short* lB = (const unsigned short*)((const char*)&ldsB[0][0] + bufOff);
        FragU aF[4], bF[2];
#pragma unroll
        for (int im = 0; im < 4; ++im) {
            int base = (wm * 64 + im * 16 + r) * LDS_STRIDE;
            aF[im].h[0] = *(const v8bf*)(lA + base + h * 8);        // K = h*8 .. h*8+7
            aF[im].h[1] = *(const v8bf*)(lA + base + 16 + h * 8);   // K = 16+h*8 ..
        }
#pragma unroll
        for (int in_ = 0; in_ < 2; ++in_) {
            int base = (wn * 32 + in_ * 16 + r) * LDS_STRIDE;
            bF[in_].h[0] = *(const v8bf*)(lB + base + h * 8);       // K = h*8 .. h*8+7
            bF[in_].h[1] = *(const v8bf*)(lB + base + 16 + h * 8);  // K = 16+h*8 ..
        }
#pragma unroll
        for (int im = 0; im < 4; ++im)
#pragma unroll
            for (int in_ = 0; in_ < 2; ++in_)
                acc[im][in_] = __builtin_amdgcn_wmma_f32_16x16x32_bf16(
                    false, aF[im].v, false, bF[in_].v,
                    (short)0, acc[im][in_], false, false);
    };

    issue(0, 0);

    const int NK = IN_K / 32;   // 128
    // Steady state: keep rolled (unroll 1) so the tied acc = wmma(..., acc)
    // stays in one fixed 8-VGPR block per tile across the backedge (no mov storm).
#pragma unroll 1
    for (int kt = 0; kt < NK - 1; ++kt) {
        wait_async0();          // this wave's copies into buf[kt&1] have landed
        __syncthreads();        // all waves landed; all reads of buf[(kt+1)&1] (iter kt-1) done
        unsigned curOff  = (unsigned)(kt & 1) * bufB;
        unsigned nextOff = curOff ^ bufB;
        issue(nextOff, kt + 1); // overlaps with compute below
        compute(curOff);
    }
    // Peeled final tile: no prefetch.
    wait_async0();
    __syncthreads();
    compute((unsigned)((NK - 1) & 1) * bufB);

    // Epilogue: D layout — VGPR v holds M = v + 8*h, lane&15 holds N.
#pragma unroll
    for (int in_ = 0; in_ < 2; ++in_) {
        int n = nBase + wn * 32 + in_ * 16 + r;
        float bv = bias[n];
#pragma unroll
        for (int im = 0; im < 4; ++im) {
            int m0 = mBase + wm * 64 + im * 16 + h * 8;
            float* op = out + (size_t)m0 * OUT_N + n;
#pragma unroll
            for (int v = 0; v < 8; ++v)
                op[(size_t)v * OUT_N] = acc[im][in_][v] + bv;
        }
    }
}

extern "C" void kernel_launch(void* const* d_in, const int* in_sizes, int n_in,
                              void* d_out, int out_size, void* d_ws, size_t ws_size,
                              hipStream_t stream) {
    const float* x       = (const float*)d_in[0];
    const float* W       = (const float*)d_in[1];
    const float* bias    = (const float*)d_in[2];
    const float* U_top   = (const float*)d_in[3];
    const float* S_top   = (const float*)d_in[4];
    const float* Vh_top  = (const float*)d_in[5];
    const float* U_tail  = (const float*)d_in[6];
    const float* S_tail  = (const float*)d_in[7];
    const float* Vh_tail = (const float*)d_in[8];
    const float* alpha   = (const float*)d_in[9];
    const float* beta    = (const float*)d_in[10];
    const float* Rm      = (const float*)d_in[11];
    float* out = (float*)d_out;

    char* ws = (char*)d_ws;
    unsigned short* xb  = (unsigned short*)ws;                                   // 128 MiB
    unsigned short* Wtb = (unsigned short*)(ws + (size_t)MROWS * IN_K * 2);      // 32 MiB
    float* RV    = (float*)(ws + (size_t)MROWS * IN_K * 2 + (size_t)OUT_N * IN_K * 2); // 1 MiB
    float* sigma = (float*)((char*)RV + (size_t)RNK * IN_K * 4);                 // 256 B

    k_sigma<<<1, 64, 0, stream>>>(S_top, alpha, beta, sigma);
    k_rv<<<dim3(IN_K / 256, RNK), 256, 0, stream>>>(Rm, Vh_tail, RV);
    k_wtilde<<<dim3(IN_K / 256, OUT_N / 16), 256, 0, stream>>>(W, U_top, U_tail, S_tail,
                                                               sigma, Vh_top, RV, Wtb);
    k_xconv<<<(MROWS * (IN_K / 4)) / 256, 256, 0, stream>>>(x, xb);
    k_gemm<<<dim3(OUT_N / 128, MROWS / 128), 256, 0, stream>>>(xb, Wtb, bias, out);
}
